// Block_41704132444810
// MI455X (gfx1250) — matrix operand order repaired and checked by reading, not compiled, not run on hardware
//
#include <hip/hip_runtime.h>
#include <hip/hip_bf16.h>
#include <math.h>

typedef __attribute__((ext_vector_type(16))) _Float16 v16h;
typedef __attribute__((ext_vector_type(8)))  float    v8f;

// ---------------------------------------------------------------------------
// Helpers
// ---------------------------------------------------------------------------
__device__ __forceinline__ v8f zero8() {
    v8f z;
#pragma unroll
    for (int i = 0; i < 8; ++i) z[i] = 0.0f;
    return z;
}

// A fragment (16x32 f16, M x K): lane l%16 = row; lanes 0-15 hold K = kHalf..kHalf+7
// and kHalf+16..kHalf+23 (kHalf = (lane>>4)*8), packed 2 halves per VGPR.
__device__ __forceinline__ v16h load_a_frag_f16(const _Float16* base, int stride,
                                                int row, int kHalf) {
    union { uint32_t u[8]; v16h v; } f;
    const _Float16* p = base + row * stride + kHalf;
#pragma unroll
    for (int j = 0; j < 4; ++j) f.u[j]     = *(const uint32_t*)(p + 2 * j);
#pragma unroll
    for (int j = 0; j < 4; ++j) f.u[4 + j] = *(const uint32_t*)(p + 16 + 2 * j);
    return f.v;
}

// A fragment built from fp32 LDS (attention probabilities), converted to f16.
__device__ __forceinline__ v16h load_a_frag_f32(const float* base, int stride,
                                                int row, int kHalf) {
    v16h a;
    const float* p = base + row * stride + kHalf;
#pragma unroll
    for (int e = 0; e < 8; ++e) {
        a[e]     = (_Float16)p[e];
        a[8 + e] = (_Float16)p[16 + e];
    }
    return a;
}

// B fragment (32x16 f16, K x N) from K-major (transposed) LDS:
// lane l%16 = column; K = kBase..kBase+7 and kBase+16..kBase+23 (kBase = (lane>>4)*8).
__device__ __forceinline__ v16h load_b_frag_f16(const _Float16* base, int stride,
                                                int col, int kBase) {
    union { uint32_t u[8]; v16h v; } f;
    const _Float16* p = base + col * stride + kBase;
#pragma unroll
    for (int j = 0; j < 8; ++j) f.u[j] = *(const uint32_t*)(p + (j >> 2) * 16 + 2 * (j & 3));
    return f.v;
}

// ---------------------------------------------------------------------------
// fp32 -> f16 cast (weights)
// ---------------------------------------------------------------------------
__global__ void cast_f32_f16(const float* __restrict__ src, _Float16* __restrict__ dst, int n) {
    for (int i = blockIdx.x * blockDim.x + threadIdx.x; i < n; i += gridDim.x * blockDim.x)
        dst[i] = (_Float16)src[i];
}

// ---------------------------------------------------------------------------
// LayerNorm: one block per row of 1024, writes f16 normalized activations
// ---------------------------------------------------------------------------
__launch_bounds__(256)
__global__ void ln_kernel(const float* __restrict__ x, const float* __restrict__ g,
                          const float* __restrict__ b, _Float16* __restrict__ out) {
    __shared__ float s1[256], s2[256];
    const int tid = threadIdx.x;
    const long row = blockIdx.x;
    const float* xr = x + row * 1024;
    float v[4], sum = 0.f, sq = 0.f;
#pragma unroll
    for (int i = 0; i < 4; ++i) {
        v[i] = xr[tid + i * 256];
        sum += v[i];
        sq  += v[i] * v[i];
    }
    s1[tid] = sum; s2[tid] = sq;
    __syncthreads();
    for (int o = 128; o > 0; o >>= 1) {
        if (tid < o) { s1[tid] += s1[tid + o]; s2[tid] += s2[tid + o]; }
        __syncthreads();
    }
    const float mu = s1[0] * (1.0f / 1024.0f);
    const float var = s2[0] * (1.0f / 1024.0f) - mu * mu;
    const float rs = rsqrtf(var + 1e-5f);
#pragma unroll
    for (int i = 0; i < 4; ++i) {
        int c = tid + i * 256;
        out[row * 1024 + c] = (_Float16)((v[i] - mu) * rs * g[c] + b[c]);
    }
}

// ---------------------------------------------------------------------------
// WMMA GEMM: C[M,N] = A[M,K](f16) * B[K,N](f16) + bias, epilogue by MODE.
//   MODE 0: scatter qkv -> q/k/v f16 arrays laid out [B,H,N,Dh]
//   MODE 1: fp32 out = resid + C   (proj / fc2 residual adds)
//   MODE 2: f16 out = gelu_exact(C)   (fc1)
// Block tile 128x128, BK=32; 8 waves in 2x4 grid, 64x32 per wave (8 WMMAs/K-step).
// A tile staged with GLOBAL_LOAD_ASYNC_TO_LDS_B128 (ASYNCcnt); B staged transposed
// via VALU (needs transpose). Next K-tile prefetched with global_prefetch_b8.
// ---------------------------------------------------------------------------
template <int MODE>
__launch_bounds__(256)
__global__ void gemm_wmma(const _Float16* __restrict__ A, const _Float16* __restrict__ Bm,
                          const float* __restrict__ bias, int K, int N,
                          const float* __restrict__ resid, float* __restrict__ outF,
                          _Float16* __restrict__ oQ, _Float16* __restrict__ oK,
                          _Float16* __restrict__ oV) {
    __shared__ _Float16 sA[128][40];   // 128 x 32 (+pad)
    __shared__ _Float16 sBt[128][36];  // B tile transposed: [N][K] (+pad)

    const int tid  = threadIdx.x;
    const int lane = tid & 31;
    const int wave = tid >> 5;
    const int wm   = wave >> 2;   // 0..1  -> 64 rows each
    const int wn   = wave & 3;    // 0..3  -> 32 cols each
    const int row0 = blockIdx.y * 128;
    const int n0   = blockIdx.x * 128;

    v8f acc[4][2];
#pragma unroll
    for (int ms = 0; ms < 4; ++ms)
#pragma unroll
        for (int ns = 0; ns < 2; ++ns) acc[ms][ns] = zero8();

    for (int kt = 0; kt < K; kt += 32) {
        // prefetch next K-tile while we compute this one
        if (kt + 32 < K) {
            if (tid < 128)
                __builtin_prefetch(A + (long)(row0 + tid) * K + kt + 32, 0, 0);
            if (tid < 64)
                __builtin_prefetch(Bm + (long)(kt + 32 + (tid >> 1)) * N + n0 + (tid & 1) * 64,
                                   0, 0);
        }

        // stage A tile asynchronously: 128 rows x 64B = 512 x 16B chunks, 2/thread
        {
            const char* Abase = (const char*)(A + (long)row0 * K + kt);
#pragma unroll
            for (int i = 0; i < 2; ++i) {
                const int e = tid + i * 256;
                const int r = e >> 2, c = e & 3;
                const uint32_t lds = (uint32_t)(uintptr_t)&sA[r][c * 8];
                const uint32_t off = (uint32_t)((r * K + c * 8) * 2);
                asm volatile("global_load_async_to_lds_b128 %0, %1, %2"
                             :: "v"(lds), "v"(off), "s"(Abase) : "memory");
            }
        }

        // stage B tile transposed: 2048 dword elements
#pragma unroll
        for (int i = 0; i < 8; ++i) {
            int e = tid + i * 256;
            int r = e >> 6, c2 = e & 63;
            uint32_t w = *(const uint32_t*)(Bm + (long)(kt + r) * N + n0 + c2 * 2);
            union { uint32_t u; _Float16 h[2]; } cv; cv.u = w;
            sBt[c2 * 2][r]     = cv.h[0];
            sBt[c2 * 2 + 1][r] = cv.h[1];
        }

        asm volatile("s_wait_asynccnt 0" ::: "memory");
        __syncthreads();

        const int kHalf = (lane >> 4) * 8;
        const int kBase = (lane >> 4) * 8;
        v16h af[4], bf[2];
#pragma unroll
        for (int ms = 0; ms < 4; ++ms)
            af[ms] = load_a_frag_f16(&sA[0][0], 40, wm * 64 + ms * 16 + (lane & 15), kHalf);
#pragma unroll
        for (int ns = 0; ns < 2; ++ns)
            bf[ns] = load_b_frag_f16(&sBt[0][0], 36, wn * 32 + ns * 16 + (lane & 15), kBase);
#pragma unroll
        for (int ms = 0; ms < 4; ++ms)
#pragma unroll
            for (int ns = 0; ns < 2; ++ns)
                acc[ms][ns] = __builtin_amdgcn_wmma_f32_16x16x32_f16(
                    false, af[ms], false, bf[ns], (short)0, acc[ms][ns], false, false);
        __syncthreads();
    }

    // Epilogue. C layout: lanes 0-15 -> N=lane, M=r; lanes 16-31 -> N=lane-16, M=8+r.
#pragma unroll
    for (int ms = 0; ms < 4; ++ms) {
#pragma unroll
        for (int ns = 0; ns < 2; ++ns) {
            const int col   = n0 + wn * 32 + ns * 16 + (lane & 15);
            const int rbase = row0 + wm * 64 + ms * 16 + ((lane >> 4) * 8);
            const float bv  = bias[col];
#pragma unroll
            for (int r = 0; r < 8; ++r) {
                const int row = rbase + r;
                const float val = acc[ms][ns][r] + bv;
                if (MODE == 0) {
                    const int b = row >> 10, n = row & 1023;
                    const int s = col >> 10, h = (col >> 6) & 15, d = col & 63;
                    const long idx = (((long)(b * 16 + h)) * 1024 + n) * 64 + d;
                    const _Float16 hv = (_Float16)val;
                    if (s == 0)      oQ[idx] = hv;
                    else if (s == 1) oK[idx] = hv;
                    else             oV[idx] = hv;
                } else if (MODE == 1) {
                    const long idx = (long)row * N + col;
                    outF[idx] = resid[idx] + val;
                } else {  // MODE 2: exact GELU -> f16
                    const float gl = 0.5f * val * (1.0f + erff(val * 0.70710678118654752f));
                    oQ[(long)row * N + col] = (_Float16)gl;
                }
            }
        }
    }
}

// ---------------------------------------------------------------------------
// Flash attention: block = (b*h, 64-row Q tile). Q/K/V f16 in [B*H, N, Dh].
// Output f16 in [B, N, H*Dh] layout for the projection GEMM.
// ---------------------------------------------------------------------------
__launch_bounds__(256)
__global__ void attn_kernel(const _Float16* __restrict__ q, const _Float16* __restrict__ k,
                            const _Float16* __restrict__ v, _Float16* __restrict__ o) {
    __shared__ _Float16 sQ[64][72];
    __shared__ _Float16 sK[64][72];
    __shared__ _Float16 sVt[64][72];   // V transposed: [Dh][seq]
    __shared__ float    sS[64][68];    // scores / probabilities (fp32)
    __shared__ float    sO[64][64];    // output accumulator (fp32)
    __shared__ float    rowM[64], rowL[64], red[64][4];

    const int  tid  = threadIdx.x;
    const int  lane = tid & 31;
    const int  wave = tid >> 5;
    const int  m0   = blockIdx.x * 64;
    const int  bh   = blockIdx.y;
    const long baseQ = (long)bh * 1024 * 64;
    const float scale = 0.125f;  // 1/sqrt(64)

    // load Q tile, init accumulators & stats
#pragma unroll
    for (int i = 0; i < 8; ++i) {
        int e = tid + i * 256;
        int r = e >> 5, d2 = e & 31;
        *(uint32_t*)&sQ[r][d2 * 2] =
            *(const uint32_t*)(q + baseQ + (long)(m0 + r) * 64 + d2 * 2);
    }
    for (int i = tid; i < 64 * 64; i += 256) sO[i >> 6][i & 63] = 0.0f;
    if (tid < 64) { rowM[tid] = -1e30f; rowL[tid] = 0.0f; }
    __syncthreads();

    for (int t = 0; t < 16; ++t) {
        // stage K tile (row-major) and V tile (transposed)
#pragma unroll
        for (int i = 0; i < 8; ++i) {
            int e = tid + i * 256;
            int r = e >> 5, d2 = e & 31;
            *(uint32_t*)&sK[r][d2 * 2] =
                *(const uint32_t*)(k + baseQ + (long)(t * 64 + r) * 64 + d2 * 2);
            uint32_t w = *(const uint32_t*)(v + baseQ + (long)(t * 64 + r) * 64 + d2 * 2);
            union { uint32_t u; _Float16 h[2]; } cv; cv.u = w;
            sVt[d2 * 2][r]     = cv.h[0];
            sVt[d2 * 2 + 1][r] = cv.h[1];
        }
        __syncthreads();

        // S = scale * Q K^T   (16 WMMA tiles / 8 waves -> 2 per wave)
#pragma unroll
        for (int ti = 0; ti < 2; ++ti) {
            const int tile = wave * 2 + ti;
            const int ms = tile >> 2, ns = tile & 3;
            v8f acc = zero8();
#pragma unroll
            for (int kk = 0; kk < 64; kk += 32) {
                v16h a = load_a_frag_f16(&sQ[0][0], 72, ms * 16 + (lane & 15),
                                         kk + ((lane >> 4) * 8));
                v16h b = load_b_frag_f16(&sK[0][0], 72, ns * 16 + (lane & 15),
                                         kk + ((lane >> 4) * 8));
                acc = __builtin_amdgcn_wmma_f32_16x16x32_f16(
                    false, a, false, b, (short)0, acc, false, false);
            }
            const int rb = ms * 16 + ((lane >> 4) * 8);
            const int cc = ns * 16 + (lane & 15);
#pragma unroll
            for (int r = 0; r < 8; ++r) sS[rb + r][cc] = acc[r] * scale;
        }
        __syncthreads();

        // online softmax update: 4 threads per row, 16 cols each
        {
            const int row = tid >> 2, seg = tid & 3, c0 = seg * 16;
            float pm = -1e30f;
#pragma unroll
            for (int c = 0; c < 16; ++c) pm = fmaxf(pm, sS[row][c0 + c]);
            red[row][seg] = pm;
            __syncthreads();
            const float mt = fmaxf(fmaxf(red[row][0], red[row][1]),
                                   fmaxf(red[row][2], red[row][3]));
            const float mold = rowM[row];
            const float mnew = fmaxf(mold, mt);
            const float alpha = expf(mold - mnew);
            float ps = 0.0f;
#pragma unroll
            for (int c = 0; c < 16; ++c) {
                float p = expf(sS[row][c0 + c] - mnew);
                sS[row][c0 + c] = p;
                ps += p;
            }
#pragma unroll
            for (int c = 0; c < 16; ++c) sO[row][c0 + c] *= alpha;
            __syncthreads();
            red[row][seg] = ps;
            __syncthreads();
            if (seg == 0) {
                rowL[row] = rowL[row] * alpha +
                            red[row][0] + red[row][1] + red[row][2] + red[row][3];
                rowM[row] = mnew;
            }
            __syncthreads();
        }

        // O += P V
#pragma unroll
        for (int ti = 0; ti < 2; ++ti) {
            const int tile = wave * 2 + ti;
            const int ms = tile >> 2, ns = tile & 3;
            v8f acc = zero8();
#pragma unroll
            for (int kk = 0; kk < 64; kk += 32) {
                v16h a = load_a_frag_f32(&sS[0][0], 68, ms * 16 + (lane & 15),
                                         kk + ((lane >> 4) * 8));
                v16h b = load_b_frag_f16(&sVt[0][0], 72, ns * 16 + (lane & 15),
                                         kk + ((lane >> 4) * 8));
                acc = __builtin_amdgcn_wmma_f32_16x16x32_f16(
                    false, a, false, b, (short)0, acc, false, false);
            }
            const int rb = ms * 16 + ((lane >> 4) * 8);
            const int cc = ns * 16 + (lane & 15);
#pragma unroll
            for (int r = 0; r < 8; ++r) sO[rb + r][cc] += acc[r];
        }
        __syncthreads();
    }

    // normalize and write out in [B, N, H*Dh] layout (f16)
    {
        const int row = tid >> 2, seg = tid & 3, c0 = seg * 16;
        const float inv = 1.0f / rowL[row];
        const int b = bh >> 4, h = bh & 15;
        const long base = ((long)(b * 1024 + m0 + row)) * 1024 + h * 64 + c0;
#pragma unroll
        for (int c = 0; c < 16; ++c) o[base + c] = (_Float16)(sO[row][c0 + c] * inv);
    }
}

// ---------------------------------------------------------------------------
// Launcher
// ---------------------------------------------------------------------------
extern "C" void kernel_launch(void* const* d_in, const int* in_sizes, int n_in,
                              void* d_out, int out_size, void* d_ws, size_t ws_size,
                              hipStream_t stream) {
    (void)in_sizes; (void)n_in; (void)out_size; (void)ws_size;

    const float* x      = (const float*)d_in[0];
    const float* ln1_g  = (const float*)d_in[1];
    const float* ln1_b  = (const float*)d_in[2];
    const float* w_qkv  = (const float*)d_in[3];
    const float* b_qkv  = (const float*)d_in[4];
    const float* w_proj = (const float*)d_in[5];
    const float* b_proj = (const float*)d_in[6];
    const float* ln2_g  = (const float*)d_in[7];
    const float* ln2_b  = (const float*)d_in[8];
    const float* w_fc1  = (const float*)d_in[9];
    const float* b_fc1  = (const float*)d_in[10];
    const float* w_fc2  = (const float*)d_in[11];
    const float* b_fc2  = (const float*)d_in[12];
    float* out = (float*)d_out;

    char* ws = (char*)d_ws;
    size_t off = 0;
    auto alloc = [&](size_t bytes) -> char* {
        char* p = ws + off;
        off += (bytes + 255) & ~(size_t)255;
        return p;
    };

    const size_t actH = 4096ull * 1024 * sizeof(_Float16);  // 8 MB per f16 activation
    _Float16* wqkv16 = (_Float16*)alloc(1024ull * 3072 * 2);
    _Float16* wproj16 = (_Float16*)alloc(1024ull * 1024 * 2);
    _Float16* wfc1_16 = (_Float16*)alloc(1024ull * 4096 * 2);
    _Float16* wfc2_16 = (_Float16*)alloc(4096ull * 1024 * 2);
    _Float16* h16     = (_Float16*)alloc(actH);     // LN output (reused for ln2)
    char*     act4    = alloc(4 * actH);            // q|k|v|attnout, reused as FFN hidden
    float*    x1      = (float*)alloc(4096ull * 1024 * 4);  // post-attention residual

    _Float16* q16  = (_Float16*)act4;
    _Float16* k16  = (_Float16*)(act4 + actH);
    _Float16* v16  = (_Float16*)(act4 + 2 * actH);
    _Float16* ao16 = (_Float16*)(act4 + 3 * actH);
    _Float16* hmid16 = (_Float16*)act4;  // 16.7M f16 = same 33.5 MB region

    // 1) weights -> f16
    cast_f32_f16<<<2048, 256, 0, stream>>>(w_qkv, wqkv16, 1024 * 3072);
    cast_f32_f16<<<2048, 256, 0, stream>>>(w_proj, wproj16, 1024 * 1024);
    cast_f32_f16<<<2048, 256, 0, stream>>>(w_fc1, wfc1_16, 1024 * 4096);
    cast_f32_f16<<<2048, 256, 0, stream>>>(w_fc2, wfc2_16, 4096 * 1024);

    // 2) LN1
    ln_kernel<<<4096, 256, 0, stream>>>(x, ln1_g, ln1_b, h16);

    // 3) qkv GEMM (4096x1024 @ 1024x3072) with scatter epilogue
    gemm_wmma<0><<<dim3(24, 32), 256, 0, stream>>>(h16, wqkv16, b_qkv, 1024, 3072,
                                                   nullptr, nullptr, q16, k16, v16);

    // 4) flash attention
    attn_kernel<<<dim3(16, 64), 256, 0, stream>>>(q16, k16, v16, ao16);

    // 5) projection GEMM + residual -> x1 (fp32)
    gemm_wmma<1><<<dim3(8, 32), 256, 0, stream>>>(ao16, wproj16, b_proj, 1024, 1024,
                                                  x, x1, nullptr, nullptr, nullptr);

    // 6) LN2
    ln_kernel<<<4096, 256, 0, stream>>>(x1, ln2_g, ln2_b, h16);

    // 7) fc1 GEMM + exact GELU -> f16 hidden
    gemm_wmma<2><<<dim3(32, 32), 256, 0, stream>>>(h16, wfc1_16, b_fc1, 1024, 4096,
                                                   nullptr, nullptr, hmid16, nullptr, nullptr);

    // 8) fc2 GEMM + residual -> d_out (fp32)
    gemm_wmma<1><<<dim3(8, 32), 256, 0, stream>>>(hmid16, wfc2_16, b_fc2, 4096, 1024,
                                                  x1, out, nullptr, nullptr, nullptr);
}
